// TransModel_26173530702095
// MI455X (gfx1250) — compile-verified, matrix operand order checked by reading, not run on hardware
//
#include <hip/hip_runtime.h>

// ---------------------------------------------------------------------------
// Model dims (fixed by the reference)
// ---------------------------------------------------------------------------
#define NB   4            // batch
#define TT   512          // seq len
#define CC   1024         // channels
#define HH   16           // heads
#define DD   64           // head dim
#define LL   8            // layers
#define FS_  64           // feature size of input
#define VV   3            // vocab
#define ROWS (NB*TT)      // 2048 token rows
#define C3   (3*CC)       // 3072
#define C4   (4*CC)       // 4096

typedef float  v8f   __attribute__((ext_vector_type(8)));
typedef __bf16 v16bf __attribute__((ext_vector_type(16)));
typedef unsigned int u32x4 __attribute__((ext_vector_type(4), aligned(16)));

union TileAB { v16bf v; u32x4 q[2]; };

static __device__ inline v8f vzero8() {
  v8f z = {0.f,0.f,0.f,0.f,0.f,0.f,0.f,0.f};
  return z;
}

// ---------------------------------------------------------------------------
// CDNA5 async global->LDS copy (16B per lane), tracked by ASYNCcnt.
// LDS byte offset = low 32 bits of the generic shared pointer (ISA 10.2:
// LDS aperture maps addr[31:0] directly).
// ---------------------------------------------------------------------------
static __device__ inline void async_copy_b128(void* lds, const void* gaddr) {
  unsigned off = (unsigned)(size_t)lds;
  asm volatile("global_load_async_to_lds_b128 %0, %1, off"
               :: "v"(off), "v"(gaddr) : "memory");
}
static __device__ inline void wait_asynccnt0() {
  asm volatile("s_wait_asynccnt 0" ::: "memory");
}

// ---------------------------------------------------------------------------
// Embedding: x[bt,c] = seqs[bt,:] @ wte_w[:,c] + wte_b[c] + wpe[t,c]   (fp32)
// ---------------------------------------------------------------------------
__global__ __launch_bounds__(256) void embed_kernel(
    const float* __restrict__ seqs, const float* __restrict__ wte_w,
    const float* __restrict__ wte_b, const float* __restrict__ wpe,
    float* __restrict__ x) {
  __shared__ float sf[FS_];
  const int bt = blockIdx.x;          // 0..2047
  const int t  = bt & (TT - 1);
  const int tid = threadIdx.x;
  if (tid < FS_) sf[tid] = seqs[(size_t)bt * FS_ + tid];
  __syncthreads();
  for (int c = tid; c < CC; c += 256) {
    float acc = wte_b[c] + wpe[(size_t)t * CC + c];
#pragma unroll 8
    for (int f = 0; f < FS_; ++f) acc += sf[f] * wte_w[(size_t)f * CC + c];
    x[(size_t)bt * CC + c] = acc;
  }
}

// ---------------------------------------------------------------------------
// LayerNorm fp32 -> bf16   (one block per row)
// ---------------------------------------------------------------------------
__global__ __launch_bounds__(256) void ln_kernel(
    const float* __restrict__ x, const float* __restrict__ sc,
    const float* __restrict__ bi, __bf16* __restrict__ out) {
  __shared__ float red[256], red2[256];
  const int row = blockIdx.x, tid = threadIdx.x;
  const float* xr = x + (size_t)row * CC;
  float s = 0.f, s2 = 0.f;
  for (int c = tid; c < CC; c += 256) { float v = xr[c]; s += v; s2 += v * v; }
  red[tid] = s; red2[tid] = s2; __syncthreads();
  for (int o = 128; o > 0; o >>= 1) {
    if (tid < o) { red[tid] += red[tid + o]; red2[tid] += red2[tid + o]; }
    __syncthreads();
  }
  const float mean = red[0] * (1.f / CC);
  const float var  = red2[0] * (1.f / CC) - mean * mean;
  const float rs   = rsqrtf(var + 1e-5f);
  __bf16* orow = out + (size_t)row * CC;
  for (int c = tid; c < CC; c += 256)
    orow[c] = (__bf16)((xr[c] - mean) * rs * sc[c] + bi[c]);
}

// ---------------------------------------------------------------------------
// Weight fp32 [K,N] -> bf16 transposed [N,K]  (64x64 LDS tiles, coalesced)
// ---------------------------------------------------------------------------
__global__ __launch_bounds__(256) void wconv_kernel(
    const float* __restrict__ W, __bf16* __restrict__ WT, int K, int N) {
  __shared__ __bf16 tile[64][65];
  const int n0 = blockIdx.x * 64;
  const int k0 = blockIdx.y * 64;
  const int tid = threadIdx.x;
#pragma unroll
  for (int i = 0; i < 16; ++i) {
    const int lin = i * 256 + tid;
    const int r = lin >> 6, c = lin & 63;           // r = k, c = n (coalesced)
    tile[r][c] = (__bf16)W[(size_t)(k0 + r) * N + n0 + c];
  }
  __syncthreads();
#pragma unroll
  for (int i = 0; i < 16; ++i) {
    const int lin = i * 256 + tid;
    const int r = lin >> 6, c = lin & 63;           // r = n, c = k (coalesced)
    WT[(size_t)(n0 + r) * K + k0 + c] = tile[c][r];
  }
}

// ---------------------------------------------------------------------------
// WMMA GEMM:  out[M,N] = A_bf16[M,K] @ WT_bf16[N,K]^T + bias
//   MODE 0: out = bf16    MODE 1: out = bf16(relu)    MODE 2: f32 += residual
// 256 threads = 8 waves; block tile 128x128; K-chunk 32.
// A and B chunks double-buffered in LDS via async global->LDS b128 copies
// (ASYNCcnt), drained with s_wait_asynccnt + barrier; operand reads are
// contiguous ds_load_b128 matching the ISA bf16 A/B register layouts.
// ---------------------------------------------------------------------------
template <int MODE>
__global__ __launch_bounds__(256) void gemm_kernel(
    const __bf16* __restrict__ A, const __bf16* __restrict__ WT,
    const float* __restrict__ bias, void* __restrict__ outp,
    const float* __restrict__ resid, int M, int K, int N) {
  __shared__ __attribute__((aligned(16))) __bf16 As[2][128][40];  // row stride 80B
  __shared__ __attribute__((aligned(16))) __bf16 Bs[2][128][40];
  const int tid  = threadIdx.x;
  const int lane = tid & 31;
  const int wave = tid >> 5;
  const int ln15 = lane & 15;
  const int ksel = ((lane >> 4) & 1) * 8;   // K sub-block select per ISA layout
  const int nb = blockIdx.x * 128;
  const int mb = blockIdx.y * 128;

  // async-stage one 128x32 bf16 chunk of A and of W^T (4 x 16B per row)
  auto issue = [&](int buf, int kc) {
#pragma unroll
    for (int i = 0; i < 2; ++i) {
      const int chunk = i * 256 + tid;          // 512 chunks of 16B
      const int row = chunk >> 2, sub = (chunk & 3) * 8;
      async_copy_b128(&As[buf][row][sub], A  + (size_t)(mb + row) * K + kc + sub);
      async_copy_b128(&Bs[buf][row][sub], WT + (size_t)(nb + row) * K + kc + sub);
    }
  };

  v8f acc[8];
#pragma unroll
  for (int t = 0; t < 8; ++t) acc[t] = vzero8();

  issue(0, 0);
  wait_asynccnt0();
  __syncthreads();

  int buf = 0;
  for (int kc = 0; kc < K; kc += 32) {
    if (kc + 32 < K) issue(buf ^ 1, kc + 32);   // overlap copies with WMMAs

    TileAB a;
    const __bf16* ap = &As[buf][wave * 16 + ln15][ksel];
    a.q[0] = *(const u32x4*)ap;
    a.q[1] = *(const u32x4*)(ap + 16);
#pragma unroll
    for (int t = 0; t < 8; ++t) {
      TileAB bt;
      const __bf16* bp = &Bs[buf][t * 16 + ln15][ksel];
      bt.q[0] = *(const u32x4*)bp;
      bt.q[1] = *(const u32x4*)(bp + 16);
      acc[t] = __builtin_amdgcn_wmma_f32_16x16x32_bf16(
          false, a.v, false, bt.v, (short)0, acc[t], false, false);
    }

    wait_asynccnt0();      // next chunk's copies have landed
    __syncthreads();       // all waves done reading `buf`
    buf ^= 1;
  }

  const int mrow = mb + wave * 16;
#pragma unroll
  for (int t = 0; t < 8; ++t) {
    const int n = nb + t * 16 + ln15;
    const float bv = bias[n];
#pragma unroll
    for (int g = 0; g < 8; ++g) {
      const size_t idx = (size_t)(mrow + g + ksel) * N + n;
      const float val = acc[t][g] + bv;
      if (MODE == 0)      ((__bf16*)outp)[idx] = (__bf16)val;
      else if (MODE == 1) ((__bf16*)outp)[idx] = (__bf16)fmaxf(val, 0.f);
      else                ((float*)outp)[idx]  = resid[idx] + val;
    }
  }
}

// ---------------------------------------------------------------------------
// Flash attention, one workgroup per (b,h), 8 waves, each wave owns 4 q-tiles
// of 16 queries.  Computes S^T = K*Q^T so the WMMA C-layout of S^T directly
// matches the B-operand layout of P^T needed for O^T = V^T * P^T (no lane
// permutes).  Online softmax state is per lane pair (m, m+16): shfl_xor(16).
// V^T staged in LDS once per block.
// ---------------------------------------------------------------------------
#define VTS 520   // 512 + 8 pad -> row stride 1040B (16B multiple)

__global__ __launch_bounds__(256) void attn_kernel(
    const __bf16* __restrict__ qkv, const int* __restrict__ seq_ls,
    __bf16* __restrict__ y) {
  extern __shared__ __bf16 VT[];                  // [64][VTS]
  const int bh = blockIdx.x;
  const int b = bh / HH, h = bh % HH;
  const int tid = threadIdx.x;

  // stage V^T : VT[d][j] = v[b,j,h,d]
  for (int idx = tid; idx < TT * DD; idx += 256) {
    const int d = idx & (DD - 1), j = idx >> 6;
    VT[d * VTS + j] = qkv[((size_t)(b * TT + j)) * C3 + 2 * CC + h * DD + d];
  }
  __syncthreads();

  const int lane = tid & 31;
  const int wave = tid >> 5;
  const int ln15 = lane & 15;
  const int ksel = ((lane >> 4) & 1) * 8;
  const int valid = seq_ls[b];
  const float scale = 0.125f;                     // 1/sqrt(64)
  const __bf16* kbase = qkv + (size_t)(b * TT) * C3 + CC + h * DD;

  for (int qt = wave; qt < TT / 16; qt += 8) {
    const int q0 = qt * 16;

    // Q^T B-operand tiles for the two d-chunks (reduction dim = d)
    TileAB bq0, bq1;
    {
      const __bf16* qp = qkv + ((size_t)(b * TT + q0 + ln15)) * C3 + h * DD;
      bq0.q[0] = *(const u32x4*)(qp + ksel);
      bq0.q[1] = *(const u32x4*)(qp + ksel + 16);
      bq1.q[0] = *(const u32x4*)(qp + 32 + ksel);
      bq1.q[1] = *(const u32x4*)(qp + 32 + ksel + 16);
    }

    v8f oacc[4];
#pragma unroll
    for (int i = 0; i < 4; ++i) oacc[i] = vzero8();
    float mrun = -3.0e38f, lrun = 0.f;

    for (int jc = 0; jc < TT; jc += 32) {
      v8f s0 = vzero8(), s1 = vzero8();
      {  // keys jc..jc+15 : S^T tile 0
        const __bf16* kp = kbase + (size_t)(jc + ln15) * C3;
        TileAB a0, a1;
        a0.q[0] = *(const u32x4*)(kp + ksel);
        a0.q[1] = *(const u32x4*)(kp + ksel + 16);
        a1.q[0] = *(const u32x4*)(kp + 32 + ksel);
        a1.q[1] = *(const u32x4*)(kp + 32 + ksel + 16);
        s0 = __builtin_amdgcn_wmma_f32_16x16x32_bf16(false, a0.v, false, bq0.v, (short)0, s0, false, false);
        s0 = __builtin_amdgcn_wmma_f32_16x16x32_bf16(false, a1.v, false, bq1.v, (short)0, s0, false, false);
      }
      {  // keys jc+16..jc+31 : S^T tile 1
        const __bf16* kp = kbase + (size_t)(jc + 16 + ln15) * C3;
        TileAB a0, a1;
        a0.q[0] = *(const u32x4*)(kp + ksel);
        a0.q[1] = *(const u32x4*)(kp + ksel + 16);
        a1.q[0] = *(const u32x4*)(kp + 32 + ksel);
        a1.q[1] = *(const u32x4*)(kp + 32 + ksel + 16);
        s1 = __builtin_amdgcn_wmma_f32_16x16x32_bf16(false, a0.v, false, bq0.v, (short)0, s1, false, false);
        s1 = __builtin_amdgcn_wmma_f32_16x16x32_bf16(false, a1.v, false, bq1.v, (short)0, s1, false, false);
      }

      // scale + key-padding mask (key index is a function of VGPR slot + lane half)
      float cm = -3.0e38f;
#pragma unroll
      for (int g = 0; g < 8; ++g) {
        const int k0i = jc + g + ksel;
        const int k1i = k0i + 16;
        const float v0 = (k0i < valid) ? s0[g] * scale : -3.0e38f;
        const float v1 = (k1i < valid) ? s1[g] * scale : -3.0e38f;
        s0[g] = v0; s1[g] = v1;
        cm = fmaxf(cm, fmaxf(v0, v1));
      }
      cm = fmaxf(cm, __shfl_xor(cm, 16, 32));
      const float mnew = fmaxf(mrun, cm);
      const float corr = __expf(mrun - mnew);
      lrun *= corr;
#pragma unroll
      for (int i = 0; i < 4; ++i) oacc[i] *= corr;

      // exp -> pack P^T directly into B-operand layout (no permute needed)
      TileAB pb;
      float psum = 0.f;
#pragma unroll
      for (int g = 0; g < 8; ++g) {
        const float p0 = __expf(s0[g] - mnew);
        const float p1 = __expf(s1[g] - mnew);
        psum += p0 + p1;
        pb.v[g]     = (__bf16)p0;   // K = ksel + g
        pb.v[8 + g] = (__bf16)p1;   // K = ksel + 16 + g
      }
      lrun += psum; mrun = mnew;

      // O^T += V^T * P^T  (4 d-tiles of 16)
#pragma unroll
      for (int dt = 0; dt < 4; ++dt) {
        TileAB av;
        const __bf16* vp = &VT[(dt * 16 + ln15) * VTS + jc + ksel];
        av.q[0] = *(const u32x4*)vp;
        av.q[1] = *(const u32x4*)(vp + 16);
        oacc[dt] = __builtin_amdgcn_wmma_f32_16x16x32_bf16(
            false, av.v, false, pb.v, (short)0, oacc[dt], false, false);
      }
    }

    const float ltot = lrun + __shfl_xor(lrun, 16, 32);
    const float inv = 1.0f / ltot;
    __bf16* yp = y + ((size_t)(b * TT + q0 + ln15)) * CC + h * DD;
#pragma unroll
    for (int dt = 0; dt < 4; ++dt)
#pragma unroll
      for (int g = 0; g < 8; ++g)
        yp[dt * 16 + g + ksel] = (__bf16)(oacc[dt][g] * inv);
  }
}

// ---------------------------------------------------------------------------
// Head: logits[row, 0..2] = lnf_bf16[row,:] @ head_w[:,3]   (fp32 out)
// ---------------------------------------------------------------------------
__global__ __launch_bounds__(256) void head_kernel(
    const __bf16* __restrict__ hbuf, const float* __restrict__ head_w,
    float* __restrict__ out) {
  __shared__ float red[VV][256];
  const int row = blockIdx.x, tid = threadIdx.x;
  float a0 = 0.f, a1 = 0.f, a2 = 0.f;
  const __bf16* hr = hbuf + (size_t)row * CC;
  for (int c = tid; c < CC; c += 256) {
    const float v = (float)hr[c];
    a0 += v * head_w[c * VV + 0];
    a1 += v * head_w[c * VV + 1];
    a2 += v * head_w[c * VV + 2];
  }
  red[0][tid] = a0; red[1][tid] = a1; red[2][tid] = a2;
  __syncthreads();
  for (int o = 128; o > 0; o >>= 1) {
    if (tid < o) {
      red[0][tid] += red[0][tid + o];
      red[1][tid] += red[1][tid + o];
      red[2][tid] += red[2][tid + o];
    }
    __syncthreads();
  }
  if (tid < VV) out[(size_t)row * VV + tid] = red[tid][0];
}

// ---------------------------------------------------------------------------
// Host driver (graph-capture safe: only kernel launches on `stream`)
// ---------------------------------------------------------------------------
extern "C" void kernel_launch(void* const* d_in, const int* in_sizes, int n_in,
                              void* d_out, int out_size, void* d_ws, size_t ws_size,
                              hipStream_t stream) {
  const float* seqs    = (const float*)d_in[0];
  const int*   seq_ls  = (const int*)  d_in[1];
  const float* wte_w   = (const float*)d_in[2];
  const float* wte_b   = (const float*)d_in[3];
  const float* wpe     = (const float*)d_in[4];
  const float* ln1_s   = (const float*)d_in[5];
  const float* ln1_b   = (const float*)d_in[6];
  const float* attn_w  = (const float*)d_in[7];
  const float* attn_b  = (const float*)d_in[8];
  const float* attnp_w = (const float*)d_in[9];
  const float* attnp_b = (const float*)d_in[10];
  const float* ln2_s   = (const float*)d_in[11];
  const float* ln2_b   = (const float*)d_in[12];
  const float* fc_w    = (const float*)d_in[13];
  const float* fc_b    = (const float*)d_in[14];
  const float* proj_w  = (const float*)d_in[15];
  const float* proj_b  = (const float*)d_in[16];
  const float* lnf_s   = (const float*)d_in[17];
  const float* lnf_b   = (const float*)d_in[18];
  const float* head_w  = (const float*)d_in[19];

  char* ws = (char*)d_ws;                               // 72 MiB used
  float*  x    = (float*) (ws);                          //  8 MiB f32 [2048,1024]
  __bf16* h    = (__bf16*)(ws + (size_t)(8)  * (1<<20)); //  4 MiB LN out
  __bf16* qkv  = (__bf16*)(ws + (size_t)(12) * (1<<20)); // 12 MiB [2048,3072]
  __bf16* yb   = (__bf16*)(ws + (size_t)(24) * (1<<20)); //  4 MiB attn out
  __bf16* fcb  = (__bf16*)(ws + (size_t)(28) * (1<<20)); // 16 MiB [2048,4096]
  __bf16* lnfb = (__bf16*)(ws + (size_t)(44) * (1<<20)); //  4 MiB final LN
  __bf16* wt_qkv  = (__bf16*)(ws + (size_t)(48) * (1<<20)); // 6 MiB [3072,1024]
  __bf16* wt_att  = (__bf16*)(ws + (size_t)(54) * (1<<20)); // 2 MiB [1024,1024]
  __bf16* wt_fc   = (__bf16*)(ws + (size_t)(56) * (1<<20)); // 8 MiB [4096,1024]
  __bf16* wt_proj = (__bf16*)(ws + (size_t)(64) * (1<<20)); // 8 MiB [1024,4096]
  float*  out  = (float*)d_out;

  embed_kernel<<<ROWS, 256, 0, stream>>>(seqs, wte_w, wte_b, wpe, x);

  for (int l = 0; l < LL; ++l) {
    // per-layer weight fp32 -> bf16 transposed (re-used scratch)
    wconv_kernel<<<dim3(C3 / 64, CC / 64), 256, 0, stream>>>(
        attn_w + (size_t)l * CC * C3, wt_qkv, CC, C3);
    wconv_kernel<<<dim3(CC / 64, CC / 64), 256, 0, stream>>>(
        attnp_w + (size_t)l * CC * CC, wt_att, CC, CC);
    wconv_kernel<<<dim3(C4 / 64, CC / 64), 256, 0, stream>>>(
        fc_w + (size_t)l * CC * C4, wt_fc, CC, C4);
    wconv_kernel<<<dim3(CC / 64, C4 / 64), 256, 0, stream>>>(
        proj_w + (size_t)l * C4 * CC, wt_proj, C4, CC);

    ln_kernel<<<ROWS, 256, 0, stream>>>(x, ln1_s + (size_t)l * CC, ln1_b + (size_t)l * CC, h);
    gemm_kernel<0><<<dim3(C3 / 128, ROWS / 128), 256, 0, stream>>>(
        h, wt_qkv, attn_b + (size_t)l * C3, qkv, nullptr, ROWS, CC, C3);
    attn_kernel<<<NB * HH, 256, (size_t)DD * VTS * sizeof(__bf16), stream>>>(qkv, seq_ls, yb);
    gemm_kernel<2><<<dim3(CC / 128, ROWS / 128), 256, 0, stream>>>(
        yb, wt_att, attnp_b + (size_t)l * CC, x, x, ROWS, CC, CC);
    ln_kernel<<<ROWS, 256, 0, stream>>>(x, ln2_s + (size_t)l * CC, ln2_b + (size_t)l * CC, h);
    gemm_kernel<1><<<dim3(C4 / 128, ROWS / 128), 256, 0, stream>>>(
        h, wt_fc, fc_b + (size_t)l * C4, fcb, nullptr, ROWS, CC, C4);
    gemm_kernel<2><<<dim3(CC / 128, ROWS / 128), 256, 0, stream>>>(
        fcb, wt_proj, proj_b + (size_t)l * CC, x, x, ROWS, C4, CC);
  }

  ln_kernel<<<ROWS, 256, 0, stream>>>(x, lnf_s, lnf_b, lnfb);
  head_kernel<<<ROWS, 256, 0, stream>>>(lnfb, head_w, out);
}
